// GATBERT_old_83829171683747
// MI455X (gfx1250) — matrix operand order repaired and checked by reading, not compiled, not run on hardware
//
#include <hip/hip_runtime.h>
#include <hip/hip_bf16.h>
#include <stdint.h>

#define NNODES 50000
#define NEDGES 800000
#define INDIM  768
#define FEAT   256      // H*F == 256 for every GAT layer
#define BATCH  128
#define SLOPE  0.2f

typedef __attribute__((ext_vector_type(8)))  unsigned short u16x8;
typedef __attribute__((ext_vector_type(16))) unsigned short u16x16;
typedef __attribute__((ext_vector_type(16))) __bf16        v16bf;
typedef __attribute__((ext_vector_type(8)))  float         v8f;

__device__ __forceinline__ unsigned short f2bf_bits(float x) {
    unsigned u = __float_as_uint(x);
    unsigned r = 0x7FFFu + ((u >> 16) & 1u);   // round-to-nearest-even
    return (unsigned short)((u + r) >> 16);
}

__device__ __forceinline__ v16bf pack_frag(u16x8 lo, u16x8 hi) {
    u16x16 u = __builtin_shufflevector(lo, hi, 0,1,2,3,4,5,6,7,8,9,10,11,12,13,14,15);
    return __builtin_bit_cast(v16bf, u);
}

// ---------------- conversion kernels ----------------

__global__ void k_f32_to_bf16(const float* __restrict__ in,
                              unsigned short* __restrict__ out, int n) {
    int i = blockIdx.x * blockDim.x + threadIdx.x;
    if (i < n) out[i] = f2bf_bits(in[i]);
}

// W [K,N] f32 row-major  ->  Wt [N,K] bf16 bits
__global__ void k_transpose_w_bf16(const float* __restrict__ W,
                                   unsigned short* __restrict__ Wt,
                                   int K, int N) {
    int k = blockIdx.x * blockDim.x + threadIdx.x;
    int n = blockIdx.y;
    if (k < K) Wt[(size_t)n * K + k] = f2bf_bits(W[(size_t)k * N + n]);
}

// ---------------- WMMA GEMM: C[M,N] = A[M,K](bf16) * Bt[N,K](bf16)^T + bias ----------------
// One wave computes a 16x64 output strip (4 accumulators share one A fragment).
// K multiple of 32; M multiple of 16; N multiple of 64.

__global__ void __launch_bounds__(32)
k_wmma_gemm_bf16(const unsigned short* __restrict__ A,
                 const unsigned short* __restrict__ Bt,
                 const float* __restrict__ bias,
                 float* __restrict__ C,
                 int N, int K, int do_relu) {
    const int lane = threadIdx.x & 31;
    const int half = lane >> 4;       // 0: lanes 0-15, 1: lanes 16-31
    const int l15  = lane & 15;

    const size_t m = (size_t)blockIdx.x * 16 + l15;   // A row owned by this lane
    const unsigned short* arow = A + m * (size_t)K;
    const int kA = half * 8;      // A frag: K = k0+kA+{0..7} and k0+kA+16+{0..7}
    const int kB = half * 8;      // B frag: K = k0+kB+{0..7} and k0+kB+16+{0..7}

    const unsigned short* brow[4];
#pragma unroll
    for (int j = 0; j < 4; ++j)
        brow[j] = Bt + ((size_t)blockIdx.y * 64 + j * 16 + l15) * (size_t)K;

    v8f acc[4] = {{}, {}, {}, {}};
    for (int k0 = 0; k0 < K; k0 += 32) {
        u16x8 a0 = *(const u16x8*)(arow + k0 + kA);
        u16x8 a1 = *(const u16x8*)(arow + k0 + kA + 16);
        v16bf av = pack_frag(a0, a1);
#pragma unroll
        for (int j = 0; j < 4; ++j) {
            u16x8 b0 = *(const u16x8*)(brow[j] + k0 + kB);
            u16x8 b1 = *(const u16x8*)(brow[j] + k0 + kB + 16);
            v16bf bv = pack_frag(b0, b1);
            acc[j] = __builtin_amdgcn_wmma_f32_16x16x32_bf16(
                         false, av, false, bv, (short)0, acc[j], false, false);
        }
    }

    // C/D layout: VGPR r -> M = r + half*8, N = l15
    const size_t mbase = (size_t)blockIdx.x * 16 + half * 8;
#pragma unroll
    for (int j = 0; j < 4; ++j) {
        const size_t n = (size_t)blockIdx.y * 64 + j * 16 + l15;
        const float bval = bias ? bias[n] : 0.0f;
#pragma unroll
        for (int r = 0; r < 8; ++r) {
            float v = acc[j][r] + bval;
            if (do_relu) v = fmaxf(v, 0.0f);
            C[(mbase + r) * (size_t)N + n] = v;
        }
    }
}

// ---------------- edge kernels ----------------
// monotonic encode/decode so unsigned atomicMax orders floats (incl. negatives)
__device__ __forceinline__ unsigned fenc(float x) {
    unsigned u = __float_as_uint(x);
    return (u & 0x80000000u) ? ~u : (u | 0x80000000u);
}
__device__ __forceinline__ float fdec(unsigned e) {
    unsigned u = (e & 0x80000000u) ? (e & 0x7FFFFFFFu) : ~e;
    return __uint_as_float(u);
}

// wave per edge: score[e,h] = dot(leaky_relu(fs[src]+fd[dst]), attn[h]); atomicMax per dst node
__global__ void k_edge_scores(const float* __restrict__ fs, const float* __restrict__ fd,
                              const int* __restrict__ src, const int* __restrict__ dst,
                              const float* __restrict__ attn,   // flat [H*F] == [256]
                              unsigned* __restrict__ nmax, float* __restrict__ exb, int H) {
    int wid  = (blockIdx.x * blockDim.x + threadIdx.x) >> 5;
    int lane = threadIdx.x & 31;
    if (wid >= NEDGES) return;
    int s = src[wid], d = dst[wid];
    const float* a = fs + (size_t)s * FEAT;
    const float* b = fd + (size_t)d * FEAT;
    int c0 = lane * 8;
    float part = 0.0f;
#pragma unroll
    for (int j = 0; j < 8; ++j) {
        float v = a[c0 + j] + b[c0 + j];
        v = v > 0.0f ? v : v * SLOPE;
        part += v * attn[c0 + j];
    }
    int lph = 32 / H;                       // lanes per head (8 or 32), wave32
    for (int m = 1; m < lph; m <<= 1) part += __shfl_xor(part, m, 32);
    if ((lane & (lph - 1)) == 0) {
        int h = lane / lph;
        exb[(size_t)wid * H + h] = part;    // raw score for now
        atomicMax(&nmax[(size_t)d * H + h], fenc(part));
    }
}

// thread per (edge,head): ex = exp(score - max[dst]); atomicAdd denom
__global__ void k_edge_exp(const int* __restrict__ dst, const unsigned* __restrict__ nmax,
                           float* __restrict__ exb, float* __restrict__ denom, int H) {
    int i = blockIdx.x * blockDim.x + threadIdx.x;
    if (i >= NEDGES * H) return;
    int e = i / H, h = i - e * H;
    int d = dst[e];
    float ex = expf(exb[i] - fdec(nmax[(size_t)d * H + h]));
    exb[i] = ex;
    atomicAdd(&denom[(size_t)d * H + h], ex);
}

// wave per edge: out[dst] += alpha * fs[src]
__global__ void k_edge_aggregate(const float* __restrict__ fs,
                                 const int* __restrict__ src, const int* __restrict__ dst,
                                 const float* __restrict__ exb, const float* __restrict__ denom,
                                 float* __restrict__ out, int H) {
    int wid  = (blockIdx.x * blockDim.x + threadIdx.x) >> 5;
    int lane = threadIdx.x & 31;
    if (wid >= NEDGES) return;
    int s = src[wid], d = dst[wid];
    int F  = FEAT / H;
    int c0 = lane * 8;
    int h  = c0 / F;                        // 8 contiguous feats always within one head
    float alpha = exb[(size_t)wid * H + h] / denom[(size_t)d * H + h];
    const float* a = fs  + (size_t)s * FEAT;
    float*       o = out + (size_t)d * FEAT;
#pragma unroll
    for (int j = 0; j < 8; ++j) atomicAdd(&o[c0 + j], alpha * a[c0 + j]);
}

// ---------------- head ----------------
// combined[b, 0:256] = mean over 4 aspects of h[idx]; combined[b, 256:1024] = pooled[b]; bf16
__global__ void k_build_combined(const float* __restrict__ h, const int* __restrict__ aidx,
                                 const float* __restrict__ pooled,
                                 unsigned short* __restrict__ comb) {
    int b = blockIdx.x;
    for (int c = threadIdx.x; c < 1024; c += blockDim.x) {
        float v;
        if (c < 256) {
            float s = 0.0f;
#pragma unroll
            for (int a = 0; a < 4; ++a) s += h[(size_t)aidx[b * 4 + a] * FEAT + c];
            v = s * 0.25f;
        } else {
            v = pooled[(size_t)b * INDIM + (c - 256)];
        }
        comb[(size_t)b * 1024 + c] = f2bf_bits(v);
    }
}

// ---------------- launch ----------------

extern "C" void kernel_launch(void* const* d_in, const int* in_sizes, int n_in,
                              void* d_out, int out_size, void* d_ws, size_t ws_size,
                              hipStream_t stream) {
    (void)in_sizes; (void)n_in; (void)out_size; (void)ws_size;

    const float* x      = (const float*)d_in[0];
    const float* pooled = (const float*)d_in[1];
    const int*   src    = (const int*)d_in[2];
    const int*   dst    = (const int*)d_in[3];
    const int*   aidx   = (const int*)d_in[4];
    const float* Ws[3]   = { (const float*)d_in[5],  (const float*)d_in[10], (const float*)d_in[15] };
    const float* bs[3]   = { (const float*)d_in[6],  (const float*)d_in[11], (const float*)d_in[16] };
    const float* Wd[3]   = { (const float*)d_in[7],  (const float*)d_in[12], (const float*)d_in[17] };
    const float* bd[3]   = { (const float*)d_in[8],  (const float*)d_in[13], (const float*)d_in[18] };
    const float* attn[3] = { (const float*)d_in[9],  (const float*)d_in[14], (const float*)d_in[19] };
    const float* Wc1 = (const float*)d_in[20]; const float* bc1 = (const float*)d_in[21];
    const float* Wc2 = (const float*)d_in[22]; const float* bc2 = (const float*)d_in[23];
    const float* Wc3 = (const float*)d_in[24]; const float* bc3 = (const float*)d_in[25];

    // workspace carve-out (256B aligned)
    size_t off = 0;
    auto carve = [&](size_t bytes) -> void* {
        void* p = (char*)d_ws + off;
        off += (bytes + 255) & ~(size_t)255;
        return p;
    };
    unsigned short* hin   = (unsigned short*)carve((size_t)NNODES * INDIM * 2); // bf16 activations
    float*          fs    = (float*)carve((size_t)NNODES * FEAT * 4);
    float*          fd    = (float*)carve((size_t)NNODES * FEAT * 4);
    float*          hout  = (float*)carve((size_t)NNODES * FEAT * 4);
    float*          exb   = (float*)carve((size_t)NEDGES * 4 * 4);
    unsigned*       nmax  = (unsigned*)carve((size_t)NNODES * 4 * 4);
    float*          denom = (float*)carve((size_t)NNODES * 4 * 4);
    unsigned short* wst   = (unsigned short*)carve((size_t)1024 * 256 * 2);
    unsigned short* wdt   = (unsigned short*)carve((size_t)768 * 256 * 2);
    unsigned short* comb  = (unsigned short*)carve((size_t)BATCH * 1024 * 2);
    float*          z1    = (float*)carve((size_t)BATCH * 256 * 4);
    unsigned short* zbf   = (unsigned short*)carve((size_t)BATCH * 256 * 2);

    const int EW_BLOCKS = (NEDGES * 32) / 256;        // wave-per-edge kernels

    auto run_layer = [&](int Kin, int li, int H) {
        k_transpose_w_bf16<<<dim3((Kin + 255) / 256, FEAT), 256, 0, stream>>>(Ws[li], wst, Kin, FEAT);
        k_transpose_w_bf16<<<dim3((Kin + 255) / 256, FEAT), 256, 0, stream>>>(Wd[li], wdt, Kin, FEAT);
        dim3 gg(NNODES / 16, FEAT / 64);
        k_wmma_gemm_bf16<<<gg, 32, 0, stream>>>(hin, wst, bs[li], fs, FEAT, Kin, 0);
        k_wmma_gemm_bf16<<<gg, 32, 0, stream>>>(hin, wdt, bd[li], fd, FEAT, Kin, 0);
        hipMemsetAsync(nmax,  0, (size_t)NNODES * H * 4, stream);
        hipMemsetAsync(denom, 0, (size_t)NNODES * H * 4, stream);
        hipMemsetAsync(hout,  0, (size_t)NNODES * FEAT * 4, stream);
        k_edge_scores   <<<EW_BLOCKS, 256, 0, stream>>>(fs, fd, src, dst, attn[li], nmax, exb, H);
        k_edge_exp      <<<(NEDGES * H + 255) / 256, 256, 0, stream>>>(dst, nmax, exb, denom, H);
        k_edge_aggregate<<<EW_BLOCKS, 256, 0, stream>>>(fs, src, dst, exb, denom, hout, H);
    };

    // layer 0 (K=768, H=4)
    k_f32_to_bf16<<<(NNODES * INDIM + 255) / 256, 256, 0, stream>>>(x, hin, NNODES * INDIM);
    run_layer(INDIM, 0, 4);
    // layer 1 (K=256, H=4)
    k_f32_to_bf16<<<(NNODES * FEAT + 255) / 256, 256, 0, stream>>>(hout, hin, NNODES * FEAT);
    run_layer(FEAT, 1, 4);
    // layer 2 (K=256, H=1)
    k_f32_to_bf16<<<(NNODES * FEAT + 255) / 256, 256, 0, stream>>>(hout, hin, NNODES * FEAT);
    run_layer(FEAT, 2, 1);

    // head MLP
    k_build_combined<<<BATCH, 256, 0, stream>>>(hout, aidx, pooled, comb);
    dim3 hg(BATCH / 16, 256 / 64);
    k_transpose_w_bf16<<<dim3(4, 256), 256, 0, stream>>>(Wc1, wst, 1024, 256);
    k_wmma_gemm_bf16<<<hg, 32, 0, stream>>>(comb, wst, bc1, z1, 256, 1024, 0);
    k_f32_to_bf16<<<(BATCH * 256 + 255) / 256, 256, 0, stream>>>(z1, zbf, BATCH * 256);
    k_transpose_w_bf16<<<dim3(1, 256), 256, 0, stream>>>(Wc2, wst, 256, 256);
    k_wmma_gemm_bf16<<<hg, 32, 0, stream>>>(zbf, wst, bc2, z1, 256, 256, 1);   // + ReLU
    k_f32_to_bf16<<<(BATCH * 256 + 255) / 256, 256, 0, stream>>>(z1, zbf, BATCH * 256);
    k_transpose_w_bf16<<<dim3(1, 256), 256, 0, stream>>>(Wc3, wst, 256, 256);
    k_wmma_gemm_bf16<<<hg, 32, 0, stream>>>(zbf, wst, bc3, (float*)d_out, 256, 256, 0);
}
